// AttentionBasic_66520453480692
// MI455X (gfx1250) — matrix operand order repaired and checked by reading, not compiled, not run on hardware
//
#include <hip/hip_runtime.h>

// ---------------------------------------------------------------------------
// Causal multi-head attention block for MI455X (gfx1250, wave32, WMMA).
//   B=2, T=2048, D=1024, NH=16, HD=64
// Pipeline (bf16 compute, f32 accumulate):
//   1) cvt:      fp32 -> bf16 copies of x, w_qkv, w_out
//   2) qkv_gemm: register-blocked WMMA GEMM (wave = 32x32 out, 4 wmma/K-step),
//                A-tile staged via global_load_async_to_lds_b128 (ASYNCcnt),
//                epilogue scatters Q/K/V into [B,NH,T,HD] bf16
//   3) flash:    per (head, 64-query block) online-softmax attention, WMMA
//                for S=Q*K^T and O+=P*V, output bf16 [B*T, D]
//   4) out_gemm: same GEMM structure, fp32 epilogue into d_out
// ---------------------------------------------------------------------------

typedef __attribute__((ext_vector_type(16))) __bf16 v16bf;
typedef __attribute__((ext_vector_type(8)))  __bf16 v8bf;
typedef __attribute__((ext_vector_type(8)))  float  v8f;

constexpr int kB  = 2;
constexpr int kT  = 2048;
constexpr int kD  = 1024;
constexpr int kNH = 16;
constexpr int kHD = 64;
constexpr int kBT = kB * kT;        // 4096 tokens
constexpr int k3D = 3 * kD;         // 3072

#define WMMA_BF16(a, b, c) \
  __builtin_amdgcn_wmma_f32_16x16x32_bf16(false, (a), false, (b), (short)0, (c), false, false)

static __device__ inline v16bf cat16(v8bf lo, v8bf hi) {
  v16bf r;
#pragma unroll
  for (int i = 0; i < 8; ++i) { r[i] = lo[i]; r[i + 8] = hi[i]; }
  return r;
}

static __device__ inline v8f vzero8() {
  v8f z;
#pragma unroll
  for (int i = 0; i < 8; ++i) z[i] = 0.0f;
  return z;
}

// Async copy of 16 bytes global -> LDS through the CDNA5 async path
// (GLOBAL_LOAD_ASYNC_TO_LDS_B128, tracked on ASYNCcnt, no VGPR data).
static __device__ inline void async_copy_b128(const void* gptr, void* lds_ptr) {
  unsigned lds_off = (unsigned)(size_t)lds_ptr;   // LDS aperture: low 32 bits
  asm volatile("global_load_async_to_lds_b128 %0, %1, off"
               :
               : "v"(lds_off), "v"(gptr)
               : "memory");
}
static __device__ inline void async_wait_all() {
  asm volatile("s_wait_asynccnt 0x0" ::: "memory");
}

// ---------------------------------------------------------------------------
// fp32 -> bf16 elementwise convert
// ---------------------------------------------------------------------------
__global__ __launch_bounds__(256) void cvt_bf16_kernel(const float* __restrict__ src,
                                                       __bf16* __restrict__ dst, int n) {
  int i = blockIdx.x * 256 + threadIdx.x;
  if (i < n) dst[i] = (__bf16)src[i];
}

// ---------------------------------------------------------------------------
// Register-blocked WMMA GEMM core used by both projections.
// 256 threads = 8 waves (2x4); block tile M=64 x N=128; wave tile 32x32.
// Per K-step(32): 2 A-frags x 2 B-frags -> 4 WMMAs.
// Returns accumulators acc[2][2] in-place via pointer.
// ---------------------------------------------------------------------------
template <int N_STRIDE>
static __device__ inline void gemm_block_64x128(
    const __bf16* __restrict__ X, const __bf16* __restrict__ W,
    int mbase, int nbase, __bf16* sA /*64*32*/, __bf16* sBT /*128*32*/,
    v8f acc[2][2]) {
  const int t    = threadIdx.x;
  const int lane = t & 31;
  const int wave = t >> 5;
  const int wm   = wave >> 2;          // 0..1
  const int wn   = wave & 3;           // 0..3
  const int ln   = lane & 15;
  const int h    = lane >> 4;

  for (int k0 = 0; k0 < kD; k0 += 32) {
    // Stage A tile 64x32 via async global->LDS (one B128 per lane)
    {
      int ar = t >> 2, ac = (t & 3) << 3;          // 64 rows x 4 chunks of 8
      async_copy_b128(&X[(size_t)(mbase + ar) * kD + k0 + ac],
                      &sA[ar * 32 + ac]);
    }
    // Stage B tile 32(k) x 128(n) transposed (coalesced 32B loads + scatter)
    {
      int kr = t >> 3, nb = (t & 7) << 4;          // 32 rows x 8 chunks of 16
      v16bf w = *(const v16bf*)&W[(size_t)(k0 + kr) * N_STRIDE + nbase + nb];
#pragma unroll
      for (int j = 0; j < 16; ++j) sBT[(nb + j) * 32 + kr] = w[j];
    }
    if (k0 + 64 <= kD) {  // gfx1250 global_prefetch_b8 for the next K slab
      __builtin_prefetch(&X[(size_t)(mbase + (t >> 2)) * kD + k0 + 32], 0, 1);
      __builtin_prefetch(&W[(size_t)(k0 + 32 + (t >> 3)) * N_STRIDE + nbase], 0, 1);
    }
    async_wait_all();       // our async LDS writes landed
    __syncthreads();        // everyone's tile writes visible

    // Fragments: 2 row tiles, 2 col tiles per wave
    v16bf am[2], bn[2];
#pragma unroll
    for (int i = 0; i < 2; ++i) {
      const int row = wm * 32 + i * 16 + ln;
      v8bf lo = *(const v8bf*)&sA[row * 32 + 16 * h];
      v8bf hi = *(const v8bf*)&sA[row * 32 + 8 + 16 * h];
      am[i] = cat16(lo, hi);
    }
#pragma unroll
    for (int j = 0; j < 2; ++j) {
      const int col = wn * 32 + j * 16 + ln;
      bn[j] = *(const v16bf*)&sBT[col * 32 + 16 * h];
    }
#pragma unroll
    for (int i = 0; i < 2; ++i)
#pragma unroll
      for (int j = 0; j < 2; ++j) acc[i][j] = WMMA_BF16(am[i], bn[j], acc[i][j]);
    __syncthreads();        // before next K-slab overwrites the tiles
  }
}

// ---------------------------------------------------------------------------
// QKV projection:  [BT, D] @ [D, 3D] + bias -> Q/K/V in [B, NH, T, HD] bf16
// ---------------------------------------------------------------------------
__global__ __launch_bounds__(256) void qkv_gemm_kernel(
    const __bf16* __restrict__ X, const __bf16* __restrict__ W,
    const float* __restrict__ bias,
    __bf16* __restrict__ Qo, __bf16* __restrict__ Ko, __bf16* __restrict__ Vo) {
  __shared__ alignas(32) __bf16 sA[64 * 32];
  __shared__ alignas(32) __bf16 sBT[128 * 32];

  const int lane = threadIdx.x & 31;
  const int wave = threadIdx.x >> 5;
  const int wm = wave >> 2, wn = wave & 3;
  const int ln = lane & 15, h = lane >> 4;
  const int mbase = blockIdx.y * 64;
  const int nbase = blockIdx.x * 128;

  v8f acc[2][2];
#pragma unroll
  for (int i = 0; i < 2; ++i)
#pragma unroll
    for (int j = 0; j < 2; ++j) acc[i][j] = vzero8();

  gemm_block_64x128<k3D>(X, W, mbase, nbase, sA, sBT, acc);

  // Epilogue: bias + scatter into per-head Q/K/V layout
#pragma unroll
  for (int j = 0; j < 2; ++j) {
    const int col  = nbase + wn * 32 + j * 16 + ln;   // 0..3071
    const float bv = bias[col];
    const int sel  = col >> 10;                       // 0:q 1:k 2:v
    const int d    = col & (kD - 1);
    const int head = d >> 6, hd = d & (kHD - 1);
    __bf16* dst = (sel == 0) ? Qo : ((sel == 1) ? Ko : Vo);
#pragma unroll
    for (int i = 0; i < 2; ++i)
#pragma unroll
      for (int r = 0; r < 8; ++r) {
        int row = mbase + wm * 32 + i * 16 + r + 8 * h;   // 0..BT-1
        int bb = row >> 11, tt = row & (kT - 1);
        dst[((size_t)(bb * kNH + head) * kT + tt) * kHD + hd] =
            (__bf16)(acc[i][j][r] + bv);
      }
  }
}

// ---------------------------------------------------------------------------
// Output projection: [BT, D] @ [D, D] + bias -> fp32 d_out
// ---------------------------------------------------------------------------
__global__ __launch_bounds__(256) void out_gemm_kernel(
    const __bf16* __restrict__ A, const __bf16* __restrict__ W,
    const float* __restrict__ bias, float* __restrict__ Out) {
  __shared__ alignas(32) __bf16 sA[64 * 32];
  __shared__ alignas(32) __bf16 sBT[128 * 32];

  const int lane = threadIdx.x & 31;
  const int wave = threadIdx.x >> 5;
  const int wm = wave >> 2, wn = wave & 3;
  const int ln = lane & 15, h = lane >> 4;
  const int mbase = blockIdx.y * 64;
  const int nbase = blockIdx.x * 128;

  v8f acc[2][2];
#pragma unroll
  for (int i = 0; i < 2; ++i)
#pragma unroll
    for (int j = 0; j < 2; ++j) acc[i][j] = vzero8();

  gemm_block_64x128<kD>(A, W, mbase, nbase, sA, sBT, acc);

#pragma unroll
  for (int j = 0; j < 2; ++j) {
    const int col  = nbase + wn * 32 + j * 16 + ln;
    const float bv = bias[col];
#pragma unroll
    for (int i = 0; i < 2; ++i)
#pragma unroll
      for (int r = 0; r < 8; ++r) {
        int row = mbase + wm * 32 + i * 16 + r + 8 * h;
        Out[(size_t)row * kD + col] = acc[i][j][r] + bv;
      }
  }
}

// ---------------------------------------------------------------------------
// Flash attention: one workgroup per (b*NH head, 64-query block), 4 waves.
// Each wave owns 16 query rows; key tiles of 32; online softmax in registers.
// ---------------------------------------------------------------------------
__global__ __launch_bounds__(128) void flash_attn_kernel(
    const __bf16* __restrict__ Q, const __bf16* __restrict__ K,
    const __bf16* __restrict__ V, __bf16* __restrict__ O) {
  __shared__ alignas(32) __bf16 sVT[64 * 32];     // V tile transposed [d][kv]
  __shared__ alignas(32) __bf16 sP[4][16 * 32];   // per-wave P tile [m][kv]

  const int t    = threadIdx.x;
  const int lane = t & 31;
  const int wave = t >> 5;
  const int ln   = lane & 15;
  const int h    = lane >> 4;
  const int bh   = blockIdx.x;                    // b*NH + head
  const int b    = bh >> 4;
  const int head = bh & (kNH - 1);

  const __bf16* Qh = Q + (size_t)bh * kT * kHD;
  const __bf16* Kh = K + (size_t)bh * kT * kHD;
  const __bf16* Vh = V + (size_t)bh * kT * kHD;

  const int q0 = blockIdx.y * 64 + wave * 16;     // this wave's first query row

  // Preload Q fragments (head dim 64 = 2 WMMA K-steps of 32)
  v16bf aq[2];
#pragma unroll
  for (int s = 0; s < 2; ++s) {
    v8bf lo = *(const v8bf*)&Qh[(size_t)(q0 + ln) * kHD + s * 32 + 16 * h];
    v8bf hi = *(const v8bf*)&Qh[(size_t)(q0 + ln) * kHD + s * 32 + 8 + 16 * h];
    aq[s] = cat16(lo, hi);
  }

  v8f o0 = vzero8(), o1 = vzero8(), o2 = vzero8(), o3 = vzero8();
  float mrow[8], lrow[8];
#pragma unroll
  for (int r = 0; r < 8; ++r) { mrow[r] = -1e30f; lrow[r] = 0.0f; }

  const int nkt = blockIdx.y * 2 + 2;             // causal: keys up to block end

  for (int kt = 0; kt < nkt; ++kt) {
    const int kb = kt * 32;

    // Stage V tile (32 keys x 64 dims) transposed into sVT
    {
      int kv = t >> 2, db = (t & 3) * 16;
      v16bf vrow = *(const v16bf*)&Vh[(size_t)(kb + kv) * kHD + db];
#pragma unroll
      for (int j = 0; j < 16; ++j) sVT[(db + j) * 32 + kv] = vrow[j];
    }
    __syncthreads();

    // S = Q * K^T for 32 keys (two 16-key column tiles), K frags from global
    v8f s0 = vzero8(), s1 = vzero8();
#pragma unroll
    for (int s = 0; s < 2; ++s) {
      v16bf b0 = *(const v16bf*)&Kh[(size_t)(kb + ln) * kHD + s * 32 + 16 * h];
      v16bf b1 = *(const v16bf*)&Kh[(size_t)(kb + 16 + ln) * kHD + s * 32 + 16 * h];
      s0 = WMMA_BF16(aq[s], b0, s0);
      s1 = WMMA_BF16(aq[s], b1, s1);
    }

    const float scale = 0.125f;                   // HD^-0.5
#pragma unroll
    for (int r = 0; r < 8; ++r) {
      const int q = q0 + r + 8 * h;
      float a0 = s0[r] * scale;
      float a1 = s1[r] * scale;
      if (kb + ln > q)      a0 = -1e30f;          // causal mask
      if (kb + 16 + ln > q) a1 = -1e30f;
      float mx = fmaxf(a0, a1);
#pragma unroll
      for (int off = 1; off < 16; off <<= 1) mx = fmaxf(mx, __shfl_xor(mx, off, 32));
      const float mnew = fmaxf(mrow[r], mx);
      const float corr = __expf(mrow[r] - mnew);
      const float p0 = __expf(a0 - mnew);
      const float p1 = __expf(a1 - mnew);
      float rs = p0 + p1;
#pragma unroll
      for (int off = 1; off < 16; off <<= 1) rs += __shfl_xor(rs, off, 32);
      lrow[r] = lrow[r] * corr + rs;
      mrow[r] = mnew;
      o0[r] *= corr; o1[r] *= corr; o2[r] *= corr; o3[r] *= corr;
      // Transpose P (C layout -> A layout) through this wave's LDS slab
      sP[wave][(r + 8 * h) * 32 + ln]      = (__bf16)p0;
      sP[wave][(r + 8 * h) * 32 + 16 + ln] = (__bf16)p1;
    }
    __syncthreads();

    // O += P * V  (K dim = 32 keys; 4 output column tiles of 16)
    v8bf plo = *(const v8bf*)&sP[wave][ln * 32 + 16 * h];
    v8bf phi = *(const v8bf*)&sP[wave][ln * 32 + 8 + 16 * h];
    v16bf pa = cat16(plo, phi);
    v16bf vb0 = *(const v16bf*)&sVT[(0 * 16 + ln) * 32 + 16 * h];
    v16bf vb1 = *(const v16bf*)&sVT[(1 * 16 + ln) * 32 + 16 * h];
    v16bf vb2 = *(const v16bf*)&sVT[(2 * 16 + ln) * 32 + 16 * h];
    v16bf vb3 = *(const v16bf*)&sVT[(3 * 16 + ln) * 32 + 16 * h];
    o0 = WMMA_BF16(pa, vb0, o0);
    o1 = WMMA_BF16(pa, vb1, o1);
    o2 = WMMA_BF16(pa, vb2, o2);
    o3 = WMMA_BF16(pa, vb3, o3);
    __syncthreads();                              // before next V tile overwrite
  }

  // Normalize and store att output as bf16 [B*T, D]
#pragma unroll
  for (int r = 0; r < 8; ++r) {
    const float inv = 1.0f / lrow[r];
    const int row = b * kT + q0 + r + 8 * h;      // global token
    const size_t base = (size_t)row * kD + head * kHD;
    O[base + 0 * 16 + ln] = (__bf16)(o0[r] * inv);
    O[base + 1 * 16 + ln] = (__bf16)(o1[r] * inv);
    O[base + 2 * 16 + ln] = (__bf16)(o2[r] * inv);
    O[base + 3 * 16 + ln] = (__bf16)(o3[r] * inv);
  }
}

// ---------------------------------------------------------------------------
// Host launcher
// ---------------------------------------------------------------------------
extern "C" void kernel_launch(void* const* d_in, const int* in_sizes, int n_in,
                              void* d_out, int out_size, void* d_ws, size_t ws_size,
                              hipStream_t stream) {
  (void)in_sizes; (void)n_in; (void)out_size; (void)ws_size;

  const float* x     = (const float*)d_in[0];  // [B, T, D]
  const float* w_qkv = (const float*)d_in[1];  // [D, 3D]
  const float* b_qkv = (const float*)d_in[2];  // [3D]
  const float* w_out = (const float*)d_in[3];  // [D, D]
  const float* b_out = (const float*)d_in[4];  // [D]
  float* out = (float*)d_out;                  // [B, T, D]

  // Carve workspace (256B-aligned slabs): ~50 MB total
  char* ws = (char*)d_ws;
  size_t off = 0;
  auto carve = [&](size_t elems) -> __bf16* {
    __bf16* p = (__bf16*)(ws + off);
    off += ((elems * sizeof(__bf16)) + 255) & ~(size_t)255;
    return p;
  };
  __bf16* xb    = carve((size_t)kBT * kD);
  __bf16* wqkvb = carve((size_t)kD * k3D);
  __bf16* woutb = carve((size_t)kD * kD);
  __bf16* Qb    = carve((size_t)kBT * kD);
  __bf16* Kb    = carve((size_t)kBT * kD);
  __bf16* Vb    = carve((size_t)kBT * kD);
  __bf16* Ab    = carve((size_t)kBT * kD);

  const int nX  = kBT * kD;      // 4,194,304
  const int nWq = kD * k3D;      // 3,145,728
  const int nWo = kD * kD;       // 1,048,576
  cvt_bf16_kernel<<<(nX  + 255) / 256, 256, 0, stream>>>(x,     xb,    nX);
  cvt_bf16_kernel<<<(nWq + 255) / 256, 256, 0, stream>>>(w_qkv, wqkvb, nWq);
  cvt_bf16_kernel<<<(nWo + 255) / 256, 256, 0, stream>>>(w_out, woutb, nWo);

  qkv_gemm_kernel<<<dim3(k3D / 128, kBT / 64), 256, 0, stream>>>(
      xb, wqkvb, b_qkv, Qb, Kb, Vb);

  flash_attn_kernel<<<dim3(kB * kNH, kT / 64), 128, 0, stream>>>(Qb, Kb, Vb, Ab);

  out_gemm_kernel<<<dim3(kD / 128, kBT / 64), 256, 0, stream>>>(
      Ab, woutb, b_out, out);
}
